// CVFR_model_87213605912854
// MI455X (gfx1250) — matrix operand order repaired and checked by reading, not compiled, not run on hardware
//
#include <hip/hip_runtime.h>
#include <hip/hip_bf16.h>
#include <stdint.h>

// ---------------------------------------------------------------------------
// CVFR attractor-network SDE, fused for MI455X (gfx1250, wave32, WMMA).
//   x: (32768, 256) f32.  50 steps of  x += DT*(-x + BETA*nl(x)@w^T) + noise.
//   Rows are fully independent -> whole 50-step loop fused in one kernel,
//   x tile resident in LDS, w^T (bf16, fragment layout) resident in LDS.
//   GEMM: v_wmma_f32_16x16x32_bf16, 4 interleaved N-tile chains per wave.
// ---------------------------------------------------------------------------

typedef __attribute__((ext_vector_type(16))) __bf16 v16bf;
typedef __attribute__((ext_vector_type(8)))  float  v8f;

#define DDIM    256
#define NCLS    10
#define NBATCH  32768
#define NSTEPS  50
#define RPB     128          // rows per block (8 waves x 16 rows)
#define XST     260          // padded LDS stride for x tile (floats)
#define GAMMA_C (1.0f/2048.0f)
#define BETA_C  0.0625f      // 1/sqrt(256)
#define DT_C    0.1f
#define SQDTSIG 0.158113883f // sqrt(0.1) * SIGMA(0.5)

// workspace layout (bytes)
#define WS_AUG 0u                       // 256 x 512 f32 augmented [phi | I]
#define WS_WT  524288u                  // 256 x 256 bf16, WMMA-fragment-major
#define WS_RA  (524288u + 131072u)      // 10 x 256 f32 (class-major r_att)
#define WS_A2  (WS_RA + 10240u)         // 10 f32 a2mean

// main-kernel dynamic LDS carve (bytes)
#define L_XS   0u                       // RPB*XST*4      = 133120
#define L_WT   133120u                  // 256*256*2      = 131072
#define L_RA   264192u                  // 10*256*4       = 10240
#define L_A2   274432u                  // 64
#define L_DV   274496u                  // RPB*4          = 512
#define L_TOT  275008u

__device__ __forceinline__ uint32_t lowbias32(uint32_t x) {
  x ^= x >> 16; x *= 0x7feb352dU;
  x ^= x >> 15; x *= 0x846ca68bU;
  x ^= x >> 16; return x;
}
// deterministic cheap N(0,1): popcount binomial (var 8) + uniform (var 1/12)
__device__ __forceinline__ float rng_normal(uint32_t idx) {
  uint32_t h  = lowbias32(idx);
  uint32_t h2 = h * 0x9E3779B9u + 0x7F4A7C15u;
  float b = (float)__popc(h) - 16.0f;
  float u = (float)(h2 >> 8) * (1.0f / 16777216.0f) - 0.5f;
  return (b + u) * 0.3517262f;          // 1/sqrt(8 + 1/12)
}
__device__ __forceinline__ float nlf(float x) {
  float x2 = x * x;
  return x2 * __builtin_amdgcn_rcpf(GAMMA_C + x2);   // v_rcp_f32, no div macro
}

// --- K1: build augmented [phi | I] --------------------------------------
__global__ void k_build_aug(const float* __restrict__ att,
                            const float* __restrict__ ptr,
                            float* __restrict__ aug) {
  int idx = blockIdx.x * 256 + threadIdx.x;     // 256*512 entries
  int row = idx >> 9, col = idx & 511;
  float v;
  if (col < 256)
    v = (col < NCLS) ? att[row * NCLS + col] : ptr[row * 246 + (col - NCLS)];
  else
    v = (col - 256 == row) ? 1.0f : 0.0f;
  aug[idx] = v;
}

// --- K2: Gauss-Jordan inverse with partial pivoting (1 block, 512 thr) ---
__global__ void k_gauss_jordan(float* __restrict__ aug) {
  const int tid = threadIdx.x;                  // 512 = one per column
  __shared__ float vals[256];
  __shared__ float prow[512];
  __shared__ float fv[256];
  __shared__ int   piv;
  for (int k = 0; k < 256; ++k) {
    if (tid < 256) vals[tid] = (tid >= k) ? fabsf(aug[tid * 512 + k]) : -1.0f;
    __syncthreads();
    if (tid == 0) {
      int p = k; float best = vals[k];
      for (int i = k + 1; i < 256; ++i) if (vals[i] > best) { best = vals[i]; p = i; }
      piv = p;
    }
    __syncthreads();
    int p = piv;
    if (p != k) {
      float a = aug[k * 512 + tid], b = aug[p * 512 + tid];
      aug[k * 512 + tid] = b; aug[p * 512 + tid] = a;
    }
    __syncthreads();
    float pv = aug[k * 512 + k];
    __syncthreads();                            // all read pv before row-k writes
    float nr = aug[k * 512 + tid] / pv;
    aug[k * 512 + tid] = nr;
    prow[tid] = nr;
    if (tid < 256) fv[tid] = (tid == k) ? 0.0f : aug[tid * 512 + k];
    __syncthreads();
    for (int i = 0; i < 256; ++i)
      aug[i * 512 + tid] -= fv[i] * prow[tid];
    __syncthreads();
  }
}

// --- K3: wT[k][j] = w[j][k] = sum_m phi[j][m]*e[m]*inv(phi)[m][k] --------
//     stored bf16 in WMMA B-fragment-major layout.
__global__ void k_compute_wT(const float* __restrict__ att,
                             const float* __restrict__ ptr,
                             const float* __restrict__ eig,
                             const float* __restrict__ aug,
                             void* __restrict__ wTv) {
  __bf16* wT = (__bf16*)wTv;
  int k = blockIdx.x, j = threadIdx.x;
  float acc = 0.0f;
  for (int m = 0; m < NCLS; ++m)
    acc += att[j * NCLS + m] * aug[m * 512 + 256 + k];
  for (int m = NCLS; m < 256; ++m)
    acc += ptr[j * 246 + (m - NCLS)] * eig[m - NCLS] * aug[m * 512 + 256 + k];
  int off = ((((k >> 5) * 16) + (j >> 4)) * 32 + (((k >> 3) & 1) * 16 + (j & 15))) * 16 + ((k & 7) | ((k >> 1) & 8));
  wT[off] = (__bf16)acc;
}

// --- K4: r_att (class-major) and a2mean ----------------------------------
__global__ void k_ratt(const float* __restrict__ att,
                       float* __restrict__ rA, float* __restrict__ a2) {
  int t = threadIdx.x;                          // 256 = one per D index
  __shared__ float red[256];
  float rr[NCLS];
  #pragma unroll
  for (int c = 0; c < NCLS; ++c) {
    float v = att[t * NCLS + c];
    float r = sqrtf(v * GAMMA_C / (1.0f - v));
    rA[c * 256 + t] = r;
    rr[c] = r * r;
  }
  for (int c = 0; c < NCLS; ++c) {
    red[t] = rr[c];
    __syncthreads();
    if (t == 0) { float s = 0.f; for (int i = 0; i < 256; ++i) s += red[i];
                  a2[c] = s * (1.0f / 256.0f); }
    __syncthreads();
  }
}

// --- K5: fused 50-step SDE loop, WMMA bf16 GEMM --------------------------
__global__ __launch_bounds__(256)
void k_cvfr_main(const float* __restrict__ xin, const void* __restrict__ wTv,
                 const float* __restrict__ rAg, const float* __restrict__ a2g,
                 float* __restrict__ out) {
  extern __shared__ char smem[];
  float*  xs   = (float*)(smem + L_XS);
  __bf16* wTs  = (__bf16*)(smem + L_WT);
  float*  rA   = (float*)(smem + L_RA);
  float*  a2m  = (float*)(smem + L_A2);
  float*  dval = (float*)(smem + L_DV);

  const int tid  = threadIdx.x;
  const int lane = tid & 31;
  const int wave = tid >> 5;
  const int half = lane >> 4;
  const int mrow = lane & 15;
  const uint32_t blockRow = blockIdx.x * RPB;

  // load x tile (coalesced f32x4 -> padded LDS rows)
  const float4* xin4 = (const float4*)(xin + (size_t)blockRow * DDIM);
  #pragma unroll
  for (int i = 0; i < 32; ++i) {
    int lin = tid + i * 256;                    // 8192 float4s
    int r = lin >> 6, c4 = lin & 63;
    *(float4*)(xs + r * XST + c4 * 4) = xin4[lin];
  }
  // load w^T fragments (straight 128KB memcpy)
  const uint4* wg4 = (const uint4*)wTv;
  uint4* wl4 = (uint4*)wTs;
  #pragma unroll
  for (int i = 0; i < 32; ++i) wl4[tid + i * 256] = wg4[tid + i * 256];
  // r_att + a2mean
  const float4* rg4 = (const float4*)rAg;
  float4* rl4 = (float4*)rA;
  for (int i = tid; i < 640; i += 256) rl4[i] = rg4[i];
  if (tid < NCLS) a2m[tid] = a2g[tid];
  __syncthreads();

  const int rowA  = wave * 16 + mrow;           // row this lane feeds into A
  const int rbase = wave * 16 + half * 8;       // first D-matrix row of this lane
  const int drow  = tid >> 1;                   // damping: 2 threads per row
  const int dhalf = tid & 1;

  #pragma unroll 1
  for (int s = 0; s < NSTEPS; ++s) {
    // ---- damping: all 256 threads, 2 per row, halves merged by shuffle ----
    {
      const float* xr = xs + drow * XST + dhalf * 128;
      float v2 = 0.f, cr[NCLS];
      #pragma unroll
      for (int c = 0; c < NCLS; ++c) cr[c] = 0.f;
      #pragma unroll 2
      for (int k = 0; k < 128; k += 4) {
        float4 xv = *(const float4*)(xr + k);
        v2 += xv.x * xv.x + xv.y * xv.y + xv.z * xv.z + xv.w * xv.w;
        #pragma unroll
        for (int c = 0; c < NCLS; ++c) {
          float4 rv = *(const float4*)(rA + c * DDIM + dhalf * 128 + k);
          cr[c] += xv.x * rv.x + xv.y * rv.y + xv.z * rv.z + xv.w * rv.w;
        }
      }
      v2 += __shfl_xor(v2, 1);
      #pragma unroll
      for (int c = 0; c < NCLS; ++c) cr[c] += __shfl_xor(cr[c], 1);
      if (dhalf == 0) {
        v2 *= (1.0f / DDIM);
        float lsum = 0.f; bool z = false;
        #pragma unroll
        for (int c = 0; c < NCLS; ++c) {
          float m = v2 - cr[c] * (2.0f / DDIM) + a2m[c];
          if (m > 0.f) lsum += __logf(m); else z = true;
        }
        float pgeo = z ? 0.0f : __expf(0.1f * lsum);
        dval[drow] = tanhf(pgeo);
      }
    }
    __syncthreads();

    // ---- build A fragments: r = nl(x) in bf16, two contiguous 8-runs/kt ----
    v16bf afrag[8];
    #pragma unroll
    for (int kt = 0; kt < 8; ++kt) {
      const float* xrp = xs + rowA * XST + kt * 32 + half * 8;
      float4 a0 = *(const float4*)(xrp);
      float4 a1 = *(const float4*)(xrp + 4);
      float4 b0 = *(const float4*)(xrp + 16);
      float4 b1 = *(const float4*)(xrp + 20);
      afrag[kt][0]  = (__bf16)nlf(a0.x); afrag[kt][1]  = (__bf16)nlf(a0.y);
      afrag[kt][2]  = (__bf16)nlf(a0.z); afrag[kt][3]  = (__bf16)nlf(a0.w);
      afrag[kt][4]  = (__bf16)nlf(a1.x); afrag[kt][5]  = (__bf16)nlf(a1.y);
      afrag[kt][6]  = (__bf16)nlf(a1.z); afrag[kt][7]  = (__bf16)nlf(a1.w);
      afrag[kt][8]  = (__bf16)nlf(b0.x); afrag[kt][9]  = (__bf16)nlf(b0.y);
      afrag[kt][10] = (__bf16)nlf(b0.z); afrag[kt][11] = (__bf16)nlf(b0.w);
      afrag[kt][12] = (__bf16)nlf(b1.x); afrag[kt][13] = (__bf16)nlf(b1.y);
      afrag[kt][14] = (__bf16)nlf(b1.z); afrag[kt][15] = (__bf16)nlf(b1.w);
    }

    // ---- hoisted per-step row constants ----
    float dloc[8];
    uint32_t rowc[8];
    const uint32_t sbase = (uint32_t)s * (uint32_t)NBATCH + blockRow + (uint32_t)rbase;
    #pragma unroll
    for (int i = 0; i < 8; ++i) {
      dloc[i] = dval[rbase + i];
      rowc[i] = (sbase + (uint32_t)i) * 256u;
    }

    // ---- 4 quads of N-tiles: four independent WMMA chains interleaved ----
    #pragma unroll 1
    for (int nq = 0; nq < 4; ++nq) {
      const int ntb = nq * 4;
      v8f acc0 = {}, acc1 = {}, acc2 = {}, acc3 = {};
      #pragma unroll
      for (int kt = 0; kt < 8; ++kt) {
        const __bf16* bp = wTs + (((kt * 16 + ntb) * 32) + lane) * 16;
        v16bf bf0 = *(const v16bf*)(bp);
        v16bf bf1 = *(const v16bf*)(bp + 512);      // +1 N-tile = 32 lanes * 16
        v16bf bf2 = *(const v16bf*)(bp + 1024);
        v16bf bf3 = *(const v16bf*)(bp + 1536);
        acc0 = __builtin_amdgcn_wmma_f32_16x16x32_bf16(
            false, afrag[kt], false, bf0, (short)0, acc0, false, false);
        acc1 = __builtin_amdgcn_wmma_f32_16x16x32_bf16(
            false, afrag[kt], false, bf1, (short)0, acc1, false, false);
        acc2 = __builtin_amdgcn_wmma_f32_16x16x32_bf16(
            false, afrag[kt], false, bf2, (short)0, acc2, false, false);
        acc3 = __builtin_amdgcn_wmma_f32_16x16x32_bf16(
            false, afrag[kt], false, bf3, (short)0, acc3, false, false);
      }
      const int col0 = (ntb + 0) * 16 + mrow;
      const int col1 = (ntb + 1) * 16 + mrow;
      const int col2 = (ntb + 2) * 16 + mrow;
      const int col3 = (ntb + 3) * 16 + mrow;
      #pragma unroll
      for (int i = 0; i < 8; ++i) {
        const int rl = rbase + i;
        float* xrow = xs + rl * XST;
        float x0 = xrow[col0], x1 = xrow[col1], x2 = xrow[col2], x3 = xrow[col3];
        float n0 = rng_normal(rowc[i] + (uint32_t)col0);
        float n1 = rng_normal(rowc[i] + (uint32_t)col1);
        float n2 = rng_normal(rowc[i] + (uint32_t)col2);
        float n3 = rng_normal(rowc[i] + (uint32_t)col3);
        float sd = SQDTSIG * dloc[i];
        xrow[col0] = x0 + DT_C * (BETA_C * acc0[i] - x0) + sd * n0;
        xrow[col1] = x1 + DT_C * (BETA_C * acc1[i] - x1) + sd * n1;
        xrow[col2] = x2 + DT_C * (BETA_C * acc2[i] - x2) + sd * n2;
        xrow[col3] = x3 + DT_C * (BETA_C * acc3[i] - x3) + sd * n3;
      }
    }
    __syncthreads();
  }

  // ---- output nl(x) ----
  float4* o4 = (float4*)(out + (size_t)blockRow * DDIM);
  #pragma unroll
  for (int i = 0; i < 32; ++i) {
    int lin = tid + i * 256;
    int r = lin >> 6, c4 = lin & 63;
    float4 v = *(const float4*)(xs + r * XST + c4 * 4);
    v.x = nlf(v.x); v.y = nlf(v.y); v.z = nlf(v.z); v.w = nlf(v.w);
    o4[lin] = v;
  }
}

// ---------------------------------------------------------------------------
extern "C" void kernel_launch(void* const* d_in, const int* in_sizes, int n_in,
                              void* d_out, int out_size, void* d_ws, size_t ws_size,
                              hipStream_t stream) {
  const float* inputs = (const float*)d_in[0];   // (32768, 256)
  const float* att    = (const float*)d_in[1];   // (256, 10)
  const float* ptr    = (const float*)d_in[2];   // (256, 246)
  const float* eig    = (const float*)d_in[3];   // (246,)
  float* out = (float*)d_out;

  char* ws = (char*)d_ws;
  float* aug = (float*)(ws + WS_AUG);
  void*  wT  = (void*)(ws + WS_WT);
  float* rA  = (float*)(ws + WS_RA);
  float* a2  = (float*)(ws + WS_A2);

  k_build_aug   <<<512, 256, 0, stream>>>(att, ptr, aug);
  k_gauss_jordan<<<1,   512, 0, stream>>>(aug);
  k_compute_wT  <<<256, 256, 0, stream>>>(att, ptr, eig, aug, wT);
  k_ratt        <<<1,   256, 0, stream>>>(att, rA, a2);
  k_cvfr_main   <<<NBATCH / RPB, 256, L_TOT, stream>>>(inputs, wT, rA, a2, out);
}
